// PointerNetwork_82746839925338
// MI455X (gfx1250) — matrix operand order repaired and checked by reading, not compiled, not run on hardware
//
#include <hip/hip_runtime.h>
#include <hip/hip_bf16.h>
#include <stdint.h>

// Problem constants (match reference)
#define BB 32
#define SS 1024
#define HH 1024
#define WW 512
#define TT 16
#define H4 4096

typedef __attribute__((ext_vector_type(16))) __bf16   v16bf;
typedef __attribute__((ext_vector_type(8)))  float    v8f;
typedef __attribute__((ext_vector_type(4)))  unsigned uint4v;

union FragU { uint4v q[2]; v16bf v; };

__device__ __forceinline__ unsigned short f2bf(float f) {
    unsigned u = __float_as_uint(f);
    u = (u + 0x7FFFu + ((u >> 16) & 1u)) >> 16;
    return (unsigned short)u;
}

__device__ __forceinline__ float sigm(float x) { return 1.0f / (1.0f + __expf(-x)); }

// ---------------------------------------------------------------------------
// fp32 -> bf16 bulk convert
// ---------------------------------------------------------------------------
__global__ void k_f2bf(unsigned short* __restrict__ dst, const float* __restrict__ src, int n) {
    int i = blockIdx.x * blockDim.x + threadIdx.x;
    if (i < n) dst[i] = f2bf(src[i]);
}

// ---------------------------------------------------------------------------
// cell0 = max_s dec, dec_in = mean_s dec (as bf16), h0 = enc[:,0,:] (as bf16)
// one thread per (b,h) column; coalesced across h
// ---------------------------------------------------------------------------
__global__ void k_init_reduce(const float* __restrict__ dec, const float* __restrict__ enc,
                              float* __restrict__ c, unsigned short* __restrict__ decin_bf,
                              unsigned short* __restrict__ h_bf) {
    int i = blockIdx.x * blockDim.x + threadIdx.x;   // B*H threads
    int b = i >> 10, h = i & 1023;
    float mx = -3.4e38f, sum = 0.0f;
    const float* col = dec + ((size_t)b * SS) * HH + h;
    for (int s = 0; s < SS; ++s) {
        float v = col[(size_t)s * HH];
        mx = fmaxf(mx, v);
        sum += v;
    }
    c[i] = mx;
    decin_bf[i] = f2bf(sum * (1.0f / (float)SS));
    h_bf[i] = f2bf(enc[((size_t)b * SS) * HH + h]);
}

// ---------------------------------------------------------------------------
// suffix sums: ssuf[b,k,:] = sum_{s>=k} dec[b,s,:], k in [0,S]; ssuf[b,S,:]=0
// ---------------------------------------------------------------------------
__global__ void k_suffix(const float* __restrict__ dec, float* __restrict__ ssuf) {
    int i = blockIdx.x * blockDim.x + threadIdx.x;   // B*H threads
    int b = i >> 10, h = i & 1023;
    size_t base = ((size_t)b * (SS + 1)) * HH + h;
    float acc = 0.0f;
    ssuf[base + (size_t)SS * HH] = 0.0f;
    const float* col = dec + ((size_t)b * SS) * HH + h;
    for (int s = SS - 1; s >= 0; --s) {
        acc += col[(size_t)s * HH];
        ssuf[base + (size_t)s * HH] = acc;
    }
}

// ---------------------------------------------------------------------------
// zero carry state (mask, bg)
// ---------------------------------------------------------------------------
__global__ void k_init_state(unsigned char* __restrict__ mask, int* __restrict__ bg) {
    int i = blockIdx.x * blockDim.x + threadIdx.x;   // B*S threads
    mask[i] = 0;
    if (i == 0) *bg = 0;
}

// ---------------------------------------------------------------------------
// WMMA GEMM: C[M,N] = A[M,K](bf16, row-major) * Wt[N,K](bf16, row-major)^T
// One wave computes a 16x64 output patch (1 M-tile x 4 N-tiles) so the A
// fragment is reused across 4 v_wmma_f32_16x16x32_bf16 issues per K-step.
// Fragment layouts (A/B k order is this kernel's own choice, shared by both; C per ISA):
//  A 16x32: lane m = lane&15; K elems {lhalf*16 + e} (K-contiguous)
//  B 32x16: lane n = lane&15; K elems {lhalf*16 + e} (K-contiguous)
//  C 16x16: VGPR j -> M = (lane>>4)*8 + j, N = lane&15
// ---------------------------------------------------------------------------
__global__ void k_wmma_gemm_nt(const unsigned short* __restrict__ A,
                               const unsigned short* __restrict__ Wt,
                               float* __restrict__ C,
                               int M, int N, int K) {
    const int wave = (blockIdx.x * blockDim.x + threadIdx.x) >> 5;
    const int lane = threadIdx.x & 31;
    const int ngroups = N >> 6;                  // 4 tiles of 16 per group
    const int total = (M >> 4) * ngroups;
    if (wave >= total) return;
    const int mt = wave / ngroups;
    const int ng = wave % ngroups;
    const int lhalf = lane >> 4;
    const int lrow  = lane & 15;

    v8f acc0 = {}, acc1 = {}, acc2 = {}, acc3 = {};

    const unsigned short* Arow = A + (size_t)(mt * 16 + lrow) * K;
    const unsigned short* B0 = Wt + (size_t)((ng * 4 + 0) * 16 + lrow) * K + lhalf * 16;
    const unsigned short* B1 = Wt + (size_t)((ng * 4 + 1) * 16 + lrow) * K + lhalf * 16;
    const unsigned short* B2 = Wt + (size_t)((ng * 4 + 2) * 16 + lrow) * K + lhalf * 16;
    const unsigned short* B3 = Wt + (size_t)((ng * 4 + 3) * 16 + lrow) * K + lhalf * 16;

    for (int kb = 0; kb < K; kb += 32) {
        FragU fa;
        fa.q[0] = *(const uint4v*)(Arow + kb + lhalf * 16);
        fa.q[1] = *(const uint4v*)(Arow + kb + lhalf * 16 + 8);
        FragU fb;
        fb.q[0] = *(const uint4v*)(B0 + kb);
        fb.q[1] = *(const uint4v*)(B0 + kb + 8);
        acc0 = __builtin_amdgcn_wmma_f32_16x16x32_bf16(false, fa.v, false, fb.v, (short)0, acc0, false, false);
        fb.q[0] = *(const uint4v*)(B1 + kb);
        fb.q[1] = *(const uint4v*)(B1 + kb + 8);
        acc1 = __builtin_amdgcn_wmma_f32_16x16x32_bf16(false, fa.v, false, fb.v, (short)0, acc1, false, false);
        fb.q[0] = *(const uint4v*)(B2 + kb);
        fb.q[1] = *(const uint4v*)(B2 + kb + 8);
        acc2 = __builtin_amdgcn_wmma_f32_16x16x32_bf16(false, fa.v, false, fb.v, (short)0, acc2, false, false);
        fb.q[0] = *(const uint4v*)(B3 + kb);
        fb.q[1] = *(const uint4v*)(B3 + kb + 8);
        acc3 = __builtin_amdgcn_wmma_f32_16x16x32_bf16(false, fa.v, false, fb.v, (short)0, acc3, false, false);
    }

    float* Cbase = C + (size_t)(mt * 16 + lhalf * 8) * N + lrow;
    const int n0 = (ng * 4) * 16;
#pragma unroll
    for (int r = 0; r < 8; ++r) {
        float* crow = Cbase + (size_t)r * N;
        crow[n0 +  0] = acc0[r];
        crow[n0 + 16] = acc1[r];
        crow[n0 + 32] = acc2[r];
        crow[n0 + 48] = acc3[r];
    }
}

// ---------------------------------------------------------------------------
// LSTM epilogue: gates(h@Whh^T) + xW(dec_in@Wih^T) + b_ih + b_hh -> c, h2(bf16)
// ---------------------------------------------------------------------------
__global__ void k_lstm_epi(const float* __restrict__ gates, const float* __restrict__ xw,
                           const float* __restrict__ b_ih, const float* __restrict__ b_hh,
                           float* __restrict__ c, unsigned short* __restrict__ h2bf) {
    int i = blockIdx.x * blockDim.x + threadIdx.x;   // B*H threads
    int b = i >> 10, h = i & 1023;
    size_t g = (size_t)b * H4;
    float gi = gates[g + h]            + xw[g + h]            + b_ih[h]            + b_hh[h];
    float gf = gates[g + HH + h]       + xw[g + HH + h]       + b_ih[HH + h]       + b_hh[HH + h];
    float gg = gates[g + 2 * HH + h]   + xw[g + 2 * HH + h]   + b_ih[2 * HH + h]   + b_hh[2 * HH + h];
    float go = gates[g + 3 * HH + h]   + xw[g + 3 * HH + h]   + b_ih[3 * HH + h]   + b_hh[3 * HH + h];
    float cn = sigm(gf) * c[i] + sigm(gi) * tanhf(gg);
    float hn = sigm(go) * tanhf(cn);
    c[i] = cn;
    h2bf[i] = f2bf(hn);
}

// ---------------------------------------------------------------------------
// scores[b,s] = sum_w tanh(blend1[b,s,w] + blend2[b,w]) * vt[w]
// one wave per (b,s) row; blend1 streams from L2 (64MB < 192MB L2)
// ---------------------------------------------------------------------------
__global__ void k_scores(const float* __restrict__ blend1, const float* __restrict__ blend2,
                         const float* __restrict__ vt, float* __restrict__ scores) {
    int row = (blockIdx.x * blockDim.x + threadIdx.x) >> 5;  // B*S rows
    int lane = threadIdx.x & 31;
    if (row >= BB * SS) return;
    int b = row >> 10;
    const float* r1 = blend1 + (size_t)row * WW;
    const float* r2 = blend2 + (size_t)b * WW;
    float sum = 0.0f;
    for (int w = lane; w < WW; w += 32)
        sum += tanhf(r1[w] + r2[w]) * vt[w];
    for (int off = 16; off > 0; off >>= 1)
        sum += __shfl_xor(sum, off, 32);
    if (lane == 0) scores[row] = sum;
}

// ---------------------------------------------------------------------------
// per-batch softmax stats: row max, log(sum exp)
// ---------------------------------------------------------------------------
__global__ void k_softmax_stats(const float* __restrict__ scores,
                                float* __restrict__ rmax, float* __restrict__ rlse) {
    __shared__ float sm[256];
    int b = blockIdx.x, tid = threadIdx.x;
    float m = -3.4e38f;
    for (int s = tid; s < SS; s += 256) m = fmaxf(m, scores[b * SS + s] + 1e-7f);
    sm[tid] = m; __syncthreads();
    for (int off = 128; off > 0; off >>= 1) {
        if (tid < off) sm[tid] = fmaxf(sm[tid], sm[tid + off]);
        __syncthreads();
    }
    float mx = sm[0]; __syncthreads();
    float sum = 0.0f;
    for (int s = tid; s < SS; s += 256) sum += __expf(scores[b * SS + s] + 1e-7f - mx);
    sm[tid] = sum; __syncthreads();
    for (int off = 128; off > 0; off >>= 1) {
        if (tid < off) sm[tid] += sm[tid + off];
        __syncthreads();
    }
    if (tid == 0) { rmax[b] = mx; rlse[b] = __logf(sm[0]); }
}

// ---------------------------------------------------------------------------
// out = -log_softmax(scores + 1e-7) with previous-mask 7.0 overwrite,
// written directly to d_out in (B, T, S) transposed layout
// ---------------------------------------------------------------------------
__global__ void k_apply_out(const float* __restrict__ scores, const float* __restrict__ rmax,
                            const float* __restrict__ rlse, const unsigned char* __restrict__ mask,
                            float* __restrict__ out, int t) {
    int i = blockIdx.x * blockDim.x + threadIdx.x;   // B*S threads
    int b = i >> 10, s = i & 1023;
    float v = rmax[b] + rlse[b] - (scores[i] + 1e-7f);
    if (mask[i]) v = 7.0f;
    out[((size_t)b * TT + t) * SS + s] = v;
}

// ---------------------------------------------------------------------------
// batch-sequential argmin scan with scalar carry bg; updates mask, kidx
// ---------------------------------------------------------------------------
__global__ void k_scan(const float* __restrict__ out, int t, unsigned char* __restrict__ mask,
                       int* __restrict__ bg_g, int* __restrict__ kidx) {
    __shared__ float sv[1024];
    __shared__ int   si[1024];
    __shared__ int s_bg, s_idx, s_valid;
    int tid = threadIdx.x;
    if (tid == 0) s_bg = *bg_g;
    __syncthreads();
    for (int b = 0; b < BB; ++b) {
        int bg = s_bg;
        float v = out[((size_t)b * TT + t) * SS + tid];
        if (tid < bg) v = 3.4e38f;                    // jnp.inf surrogate
        sv[tid] = v; si[tid] = tid;
        __syncthreads();
        for (int off = 512; off > 0; off >>= 1) {
            if (tid < off) {
                float v2 = sv[tid + off]; int i2 = si[tid + off];
                if (v2 < sv[tid] || (v2 == sv[tid] && i2 < si[tid])) { sv[tid] = v2; si[tid] = i2; }
            }
            __syncthreads();
        }
        if (tid == 0) {
            int abs_idx = si[0];
            int valid = (bg < SS) ? 1 : 0;
            s_idx = abs_idx; s_valid = valid;
            if (valid) s_bg = abs_idx - bg + 1;
            kidx[b] = valid ? abs_idx : SS;           // SS -> h_new = 0 (ssuf[S]=0)
        }
        __syncthreads();
        mask[b * SS + tid] = s_valid ? (unsigned char)(tid < s_idx) : (unsigned char)1;
        __syncthreads();
    }
    if (tid == 0) *bg_g = s_bg;
}

// ---------------------------------------------------------------------------
// h_new[b,:] = ssuf[b, kidx[b], :] / S  (as bf16 for next gates GEMM)
// ---------------------------------------------------------------------------
__global__ void k_hnew(const float* __restrict__ ssuf, const int* __restrict__ kidx,
                       unsigned short* __restrict__ hbf) {
    int i = blockIdx.x * blockDim.x + threadIdx.x;   // B*H threads
    int b = i >> 10, h = i & 1023;
    int k = kidx[b];
    float v = ssuf[((size_t)b * (SS + 1) + k) * HH + h] * (1.0f / (float)SS);
    hbf[i] = f2bf(v);
}

// ---------------------------------------------------------------------------
extern "C" void kernel_launch(void* const* d_in, const int* in_sizes, int n_in,
                              void* d_out, int out_size, void* d_ws, size_t ws_size,
                              hipStream_t stream) {
    const float* enc  = (const float*)d_in[0];
    const float* dec  = (const float*)d_in[1];
    const float* Wih  = (const float*)d_in[2];
    const float* Whh  = (const float*)d_in[3];
    const float* b_ih = (const float*)d_in[4];
    const float* b_hh = (const float*)d_in[5];
    const float* W1   = (const float*)d_in[6];
    const float* W2   = (const float*)d_in[7];
    const float* vt   = (const float*)d_in[8];
    float* out = (float*)d_out;

    // workspace carve-out (256B aligned)
    char* w = (char*)d_ws;
    size_t off = 0;
    auto carve = [&](size_t bytes) { void* p = w + off; off = (off + bytes + 255) & ~(size_t)255; return p; };
    unsigned short* dec_bf   = (unsigned short*)carve((size_t)BB * SS * HH * 2);
    float*          ssuf     = (float*)carve((size_t)BB * (SS + 1) * HH * 4);
    float*          blend1   = (float*)carve((size_t)BB * SS * WW * 4);
    unsigned short* w1_bf    = (unsigned short*)carve((size_t)WW * HH * 2);
    unsigned short* w2_bf    = (unsigned short*)carve((size_t)WW * HH * 2);
    unsigned short* wih_bf   = (unsigned short*)carve((size_t)H4 * HH * 2);
    unsigned short* whh_bf   = (unsigned short*)carve((size_t)H4 * HH * 2);
    unsigned short* decin_bf = (unsigned short*)carve((size_t)BB * HH * 2);
    unsigned short* h_bf     = (unsigned short*)carve((size_t)BB * HH * 2);
    unsigned short* h2_bf    = (unsigned short*)carve((size_t)BB * HH * 2);
    float*          c_st     = (float*)carve((size_t)BB * HH * 4);
    float*          xw       = (float*)carve((size_t)BB * H4 * 4);
    float*          gates    = (float*)carve((size_t)BB * H4 * 4);
    float*          blend2   = (float*)carve((size_t)BB * WW * 4);
    float*          scores   = (float*)carve((size_t)BB * SS * 4);
    float*          rmax     = (float*)carve(BB * 4);
    float*          rlse     = (float*)carve(BB * 4);
    unsigned char*  mask     = (unsigned char*)carve((size_t)BB * SS);
    int*            kidx     = (int*)carve(BB * 4);
    int*            bg       = (int*)carve(4);

    // --- one-time preprocessing ---
    k_f2bf<<<(BB * SS * HH + 255) / 256, 256, 0, stream>>>(dec_bf, dec, BB * SS * HH);
    k_f2bf<<<(WW * HH + 255) / 256, 256, 0, stream>>>(w1_bf, W1, WW * HH);
    k_f2bf<<<(WW * HH + 255) / 256, 256, 0, stream>>>(w2_bf, W2, WW * HH);
    k_f2bf<<<(H4 * HH + 255) / 256, 256, 0, stream>>>(wih_bf, Wih, H4 * HH);
    k_f2bf<<<(H4 * HH + 255) / 256, 256, 0, stream>>>(whh_bf, Whh, H4 * HH);
    k_init_reduce<<<(BB * HH) / 256, 256, 0, stream>>>(dec, enc, c_st, decin_bf, h_bf);
    k_suffix<<<(BB * HH) / 256, 256, 0, stream>>>(dec, ssuf);
    k_init_state<<<(BB * SS) / 256, 256, 0, stream>>>(mask, bg);

    // xW = dec_in @ Wih^T  (M=32, N=4096, K=1024) -> 128 waves -> 16 blocks
    k_wmma_gemm_nt<<<16, 256, 0, stream>>>(decin_bf, wih_bf, xw, BB, H4, HH);
    // blend1 = dec @ W1^T  (M=32768, N=512, K=1024) -> 16384 waves -> 2048 blocks
    k_wmma_gemm_nt<<<2048, 256, 0, stream>>>(dec_bf, w1_bf, blend1, BB * SS, WW, HH);

    // --- T recurrent steps ---
    for (int t = 0; t < TT; ++t) {
        k_wmma_gemm_nt<<<16, 256, 0, stream>>>(h_bf, whh_bf, gates, BB, H4, HH);
        k_lstm_epi<<<(BB * HH) / 256, 256, 0, stream>>>(gates, xw, b_ih, b_hh, c_st, h2_bf);
        k_wmma_gemm_nt<<<2, 256, 0, stream>>>(h2_bf, w2_bf, blend2, BB, WW, HH);
        k_scores<<<(BB * SS * 32) / 256, 256, 0, stream>>>(blend1, blend2, vt, scores);
        k_softmax_stats<<<BB, 256, 0, stream>>>(scores, rmax, rlse);
        k_apply_out<<<(BB * SS) / 256, 256, 0, stream>>>(scores, rmax, rlse, mask, out, t);
        k_scan<<<1, 1024, 0, stream>>>(out, t, mask, bg, kidx);
        k_hnew<<<(BB * HH) / 256, 256, 0, stream>>>(ssuf, kidx, h_bf);
    }
}
